// Attention_75960791598018
// MI455X (gfx1250) — matrix operand order repaired and checked by reading, not compiled, not run on hardware
//
#include <hip/hip_runtime.h>
#include <hip/hip_bf16.h>

typedef __bf16 bf16;
typedef __attribute__((ext_vector_type(16))) __bf16       v16bf;
typedef __attribute__((ext_vector_type(8)))  float        v8f;
typedef __attribute__((ext_vector_type(4)))  unsigned int u32x4;

#define D_MODEL 1024
#define SEQ     2048
#define NHEADS  4
#define DHEAD   256
#define BATCH   8
#define MTOT    (BATCH*SEQ)   /* 16384 */

#define KSTR 264   /* LDS row stride (elems) for K tile: 528B -> conflict-free b128 */
#define VSTR 40    /* LDS row stride (elems) for V tile:  80B -> conflict-free b128 */

enum GemmMode { OUT_F32 = 0, OUT_QK = 1, OUT_VT = 2 };

union FragBF {
    v16bf v;
    u32x4 u[2];
};

__device__ __forceinline__ v8f vzero8() {
    v8f z;
#pragma unroll
    for (int i = 0; i < 8; ++i) z[i] = 0.0f;
    return z;
}

__device__ __forceinline__ v8f wmma_bf16(v16bf a, v16bf b, v8f c) {
    // D = A(16x32 bf16) x B(32x16 bf16) + C(16x16 f32)
    return __builtin_amdgcn_wmma_f32_16x16x32_bf16(
        /*neg_a=*/false, a, /*neg_b=*/false, b,
        /*c_mod=*/(short)0, c, /*reuse_a=*/false, /*reuse_b=*/false);
}

// reductions across the 16-lane half that shares a C-fragment row (wave32)
__device__ __forceinline__ float half_max16(float x) {
#pragma unroll
    for (int m = 1; m < 16; m <<= 1) x = fmaxf(x, __shfl_xor(x, m, 32));
    return x;
}
__device__ __forceinline__ float half_sum16(float x) {
#pragma unroll
    for (int m = 1; m < 16; m <<= 1) x += __shfl_xor(x, m, 32);
    return x;
}

// CDNA5 async global -> LDS copy (16B per lane), tracked by ASYNCcnt
__device__ __forceinline__ void async_ld_b128(unsigned lds_off, const void* gptr) {
    unsigned long long ga = (unsigned long long)gptr;
    asm volatile("global_load_async_to_lds_b128 %0, %1, off"
                 :: "v"(lds_off), "v"(ga) : "memory");
}
__device__ __forceinline__ void wait_async0() {
    asm volatile("s_wait_asynccnt 0x0" ::: "memory");
}

// ---------------------------------------------------------------- converters

__global__ void cvt_f32_to_bf16(const float* __restrict__ in, bf16* __restrict__ out, int n) {
    int i = blockIdx.x * 256 + threadIdx.x;
    if (i < n) out[i] = (bf16)in[i];
}

// W_{Q,K,V}: in [H=4][D=1024][Dh=256] f32 -> out bf16 [(h*256+e)][d]  (Bt layout: n-major, k contiguous)
__global__ void transpose_wqkv(const float* __restrict__ in, bf16* __restrict__ out) {
    int o = blockIdx.x * 256 + threadIdx.x;      // 4*1024*256 = 1M total
    int d = o & (D_MODEL - 1);
    int r = o >> 10;                              // r = h*256 + e
    int h = r >> 8;
    int e = r & (DHEAD - 1);
    out[o] = (bf16)in[((size_t)h * D_MODEL + d) * DHEAD + e];
}

// W_O flattens to [K=1024][N=1024] row-major; produce Bt[n][k]
__global__ void transpose_wo(const float* __restrict__ in, bf16* __restrict__ out) {
    int o = blockIdx.x * 256 + threadIdx.x;
    int k = o & (D_MODEL - 1);
    int n = o >> 10;
    out[o] = (bf16)in[(size_t)k * D_MODEL + n];
}

// ---------------------------------------------------------------- WMMA GEMM
// C[M,N] = A[M,K](bf16) * Bt[N,K](bf16)^T, + bias[n], * scale, stored per `mode`.
// Block: 256 threads = 8 waves; block tile 128x128; wave tile 32x64 (2x4 frags).

__global__ void __launch_bounds__(256, 2)
gemm_bf16(const bf16* __restrict__ A, const bf16* __restrict__ Bt,
          const float* __restrict__ bias, void* __restrict__ outp,
          int M, int N, int K, int mode, float scale)
{
    const int lane = threadIdx.x & 31;
    const int wid  = threadIdx.x >> 5;
    const int hf   = lane >> 4;       // which 16-lane half
    const int l16  = lane & 15;
    const int m0 = blockIdx.y * 128 + (wid & 3) * 32;
    const int n0 = blockIdx.x * 128 + (wid >> 2) * 64;

    v8f acc[2][4];
#pragma unroll
    for (int i = 0; i < 2; ++i)
#pragma unroll
        for (int j = 0; j < 4; ++j) acc[i][j] = vzero8();

    for (int kk = 0; kk < K; kk += 32) {
        FragBF a[2], b[4];
#pragma unroll
        for (int i = 0; i < 2; ++i) {
            const bf16* ap = A + (size_t)(m0 + i * 16 + l16) * K + kk + hf * 8;
            a[i].u[0] = *(const u32x4*)ap;          // K = hf*8 .. +7
            a[i].u[1] = *(const u32x4*)(ap + 16);   // K = 16+hf*8 .. +7
        }
#pragma unroll
        for (int j = 0; j < 4; ++j) {
            const bf16* bp = Bt + (size_t)(n0 + j * 16 + l16) * K + kk + hf * 8;
            b[j].u[0] = *(const u32x4*)bp;          // K = hf*8 .. +7
            b[j].u[1] = *(const u32x4*)(bp + 16);   // K = 16+hf*8 .. +7
        }
#pragma unroll
        for (int i = 0; i < 2; ++i)
#pragma unroll
            for (int j = 0; j < 4; ++j)
                acc[i][j] = wmma_bf16(a[i].v, b[j].v, acc[i][j]);
    }

#pragma unroll
    for (int i = 0; i < 2; ++i) {
#pragma unroll
        for (int j = 0; j < 4; ++j) {
#pragma unroll
            for (int v = 0; v < 8; ++v) {
                int m = m0 + i * 16 + v + 8 * hf;
                int n = n0 + j * 16 + l16;
                float val = acc[i][j][v];
                if (bias) val += bias[n];
                val *= scale;
                if (mode == OUT_F32) {
                    ((float*)outp)[(size_t)m * N + n] = val;
                } else {
                    int bb = m / SEQ, s = m & (SEQ - 1);
                    int h  = n / DHEAD, e = n & (DHEAD - 1);
                    if (mode == OUT_QK)   // [B,H,S,Dh]
                        ((bf16*)outp)[(((size_t)bb * NHEADS + h) * SEQ + s) * DHEAD + e] = (bf16)val;
                    else                  // OUT_VT: [B,H,Dh,S]
                        ((bf16*)outp)[(((size_t)bb * NHEADS + h) * DHEAD + e) * SEQ + s] = (bf16)val;
                }
            }
        }
    }
}

// ------------------------------------------------------- flash attention core
// One wave per 16-row Q tile; 4 waves/block share (b,h) so K/V tiles are staged
// into LDS once per block via async global->LDS copies, double buffered.
// Q:[B,H,S,Dh], K:[B,H,S,Dh], Vt:[B,H,Dh,S], Z:[B,S,H*Dh] bf16.
// Scale 1/sqrt(Dh) is pre-folded into Q.

__device__ __forceinline__ void stage_kv(const bf16* __restrict__ Kb,
                                         const bf16* __restrict__ Vb,
                                         int sc, bf16* kbuf, bf16* vbuf, int tid)
{
    // K tile: 32 rows x first 128 of 256 elems (8KB) -> 512 x 16B chunks
    const bf16* kg = Kb + (size_t)sc * DHEAD;
#pragma unroll
    for (int i = 0; i < 4; ++i) {
        int c = tid + i * 128;                 // 0..511
        int row = c >> 4, intra = c & 15;
        unsigned lofs = (unsigned)(size_t)(kbuf + row * KSTR + intra * 8);
        async_ld_b128(lofs, kg + row * DHEAD + intra * 8);
    }
    // V tile: 256 rows x 32 elems (global row stride SEQ) -> 1024 x 16B chunks
#pragma unroll
    for (int i = 0; i < 8; ++i) {
        int c = tid + i * 128;                 // 0..1023
        int row = c >> 2, intra = c & 3;
        unsigned lofs = (unsigned)(size_t)(vbuf + row * VSTR + intra * 8);
        async_ld_b128(lofs, Vb + (size_t)row * SEQ + sc + intra * 8);
    }
}

__global__ void __launch_bounds__(128, 1)
attn_fa(const bf16* __restrict__ Q, const bf16* __restrict__ Kx,
        const bf16* __restrict__ Vt, bf16* __restrict__ Z)
{
    __shared__ bf16 Ksh[2][32 * KSTR];       // 33,792 B
    __shared__ bf16 Vsh[2][256 * VSTR];      // 40,960 B
    __shared__ bf16 psh[4][16][32];          //  4,096 B

    const int tid  = threadIdx.x;
    const int lane = tid & 31;
    const int wid  = tid >> 5;
    const int hf   = lane >> 4;
    const int l16  = lane & 15;

    const int bh = blockIdx.x >> 5;          // 32 blocks per (b,h)
    const int qg = blockIdx.x & 31;
    const int b  = bh >> 2;
    const int h  = bh & 3;
    const int q0 = (qg * 4 + wid) * 16;

    const bf16* Qb = Q  + ((size_t)bh * SEQ + q0) * DHEAD;
    const bf16* Kb = Kx + (size_t)bh * SEQ * DHEAD;
    const bf16* Vb = Vt + (size_t)bh * DHEAD * SEQ;

    // resident Q fragments: 8 x (16x32) covering Dh=256
    FragBF qf[8];
#pragma unroll
    for (int kk = 0; kk < 8; ++kk) {
        const bf16* qp = Qb + (size_t)l16 * DHEAD + kk * 32 + hf * 8;
        qf[kk].u[0] = *(const u32x4*)qp;
        qf[kk].u[1] = *(const u32x4*)(qp + 16);
    }

    v8f oacc[16];
#pragma unroll
    for (int nf = 0; nf < 16; ++nf) oacc[nf] = vzero8();
    float mrun[8], lrun[8];
#pragma unroll
    for (int v = 0; v < 8; ++v) { mrun[v] = -1e30f; lrun[v] = 0.0f; }

    stage_kv(Kb, Vb, 0, Ksh[0], Vsh[0], tid);     // prologue: tile 0

    const int NT = SEQ / 32;                      // 64 k-tiles
    for (int kt = 0; kt < NT; ++kt) {
        wait_async0();                            // this wave's stage done
        __syncthreads();                          // all waves' stages done
        if (kt + 1 < NT)
            stage_kv(Kb, Vb, (kt + 1) * 32, Ksh[(kt + 1) & 1], Vsh[(kt + 1) & 1], tid);

        const bf16* kbuf = Ksh[kt & 1];
        const bf16* vbuf = Vsh[kt & 1];

        v8f s0 = vzero8(), s1 = vzero8();
#pragma unroll
        for (int kk = 0; kk < 8; ++kk) {
            FragBF k0, k1;
            const bf16* kp0 = kbuf + (size_t)l16 * KSTR + kk * 32 + hf * 8;
            const bf16* kp1 = kp0 + 16 * KSTR;
            k0.u[0] = *(const u32x4*)kp0; k0.u[1] = *(const u32x4*)(kp0 + 16);
            k1.u[0] = *(const u32x4*)kp1; k1.u[1] = *(const u32x4*)(kp1 + 16);
            s0 = wmma_bf16(qf[kk].v, k0.v, s0);
            s1 = wmma_bf16(qf[kk].v, k1.v, s1);
        }

        // online softmax update (C layout: row = v + 8*hf, col = l16)
        float alpha[8];
#pragma unroll
        for (int v = 0; v < 8; ++v) {
            float mx = half_max16(fmaxf(s0[v], s1[v]));
            float mn = fmaxf(mrun[v], mx);
            alpha[v] = __expf(mrun[v] - mn);
            mrun[v]  = mn;
            float p0 = __expf(s0[v] - mn);
            float p1 = __expf(s1[v] - mn);
            lrun[v]  = lrun[v] * alpha[v] + half_sum16(p0 + p1);
            int r = v + 8 * hf;
            psh[wid][r][l16]      = (bf16)p0;
            psh[wid][r][l16 + 16] = (bf16)p1;
        }
#pragma unroll
        for (int nf = 0; nf < 16; ++nf)
#pragma unroll
            for (int v = 0; v < 8; ++v) oacc[nf][v] *= alpha[v];

        __syncthreads();   // P tile visible (uniform trip count across waves)

        // reload P as an A fragment (16x32 bf16)
        FragBF pf;
        const bf16* pp = &psh[wid][l16][hf * 8];
        pf.u[0] = *(const u32x4*)pp;
        pf.u[1] = *(const u32x4*)(pp + 16);

        // O += P * V (V tile rows contiguous along s in LDS)
#pragma unroll
        for (int nf = 0; nf < 16; ++nf) {
            FragBF vf;
            const bf16* vp = vbuf + (nf * 16 + l16) * VSTR + hf * 8;
            vf.u[0] = *(const u32x4*)vp;
            vf.u[1] = *(const u32x4*)(vp + 16);
            oacc[nf] = wmma_bf16(pf.v, vf.v, oacc[nf]);
        }
        __syncthreads();
    }

#pragma unroll
    for (int v = 0; v < 8; ++v) lrun[v] = 1.0f / lrun[v];
#pragma unroll
    for (int nf = 0; nf < 16; ++nf) {
#pragma unroll
        for (int v = 0; v < 8; ++v) {
            int r = v + 8 * hf;
            size_t idx = ((size_t)b * SEQ + q0 + r) * D_MODEL + h * DHEAD + nf * 16 + l16;
            Z[idx] = (bf16)(oacc[nf][v] * lrun[v]);
        }
    }
}

// ---------------------------------------------------------------------- host

extern "C" void kernel_launch(void* const* d_in, const int* in_sizes, int n_in,
                              void* d_out, int out_size, void* d_ws, size_t ws_size,
                              hipStream_t stream)
{
    const float* x   = (const float*)d_in[0];
    const float* W_Q = (const float*)d_in[1];
    const float* b_Q = (const float*)d_in[2];
    const float* W_K = (const float*)d_in[3];
    const float* b_K = (const float*)d_in[4];
    const float* W_V = (const float*)d_in[5];
    const float* b_V = (const float*)d_in[6];
    const float* W_O = (const float*)d_in[7];
    const float* b_O = (const float*)d_in[8];

    const size_t MK = (size_t)MTOT * D_MODEL;      // 16M elements
    const size_t WW = (size_t)D_MODEL * D_MODEL;   // 1M elements
    bf16* Xb  = (bf16*)d_ws;
    bf16* WtQ = Xb  + MK;
    bf16* WtK = WtQ + WW;
    bf16* WtV = WtK + WW;
    bf16* WtO = WtV + WW;
    bf16* Qb  = WtO + WW;
    bf16* Kb  = Qb  + MK;
    bf16* Vtb = Kb  + MK;
    bf16* Zb  = Vtb + MK;

    cvt_f32_to_bf16<<<(int)(MK / 256), 256, 0, stream>>>(x, Xb, (int)MK);
    transpose_wqkv<<<(int)(WW / 256), 256, 0, stream>>>(W_Q, WtQ);
    transpose_wqkv<<<(int)(WW / 256), 256, 0, stream>>>(W_K, WtK);
    transpose_wqkv<<<(int)(WW / 256), 256, 0, stream>>>(W_V, WtV);
    transpose_wo  <<<(int)(WW / 256), 256, 0, stream>>>(W_O, WtO);

    dim3 g(D_MODEL / 128, MTOT / 128);
    const float sc = 0.0625f;  // 1/sqrt(DHEAD)
    gemm_bf16<<<g, 256, 0, stream>>>(Xb, WtQ, b_Q, Qb,  MTOT, D_MODEL, D_MODEL, OUT_QK, sc);
    gemm_bf16<<<g, 256, 0, stream>>>(Xb, WtK, b_K, Kb,  MTOT, D_MODEL, D_MODEL, OUT_QK, 1.0f);
    gemm_bf16<<<g, 256, 0, stream>>>(Xb, WtV, b_V, Vtb, MTOT, D_MODEL, D_MODEL, OUT_VT, 1.0f);

    attn_fa<<<BATCH * NHEADS * (SEQ / 64), 128, 0, stream>>>(Qb, Kb, Vtb, Zb);

    gemm_bf16<<<g, 256, 0, stream>>>(Zb, WtO, b_O, d_out, MTOT, D_MODEL, D_MODEL, OUT_F32, 1.0f);
}
